// RagatInteractE_70196945486560
// MI455X (gfx1250) — matrix operand order repaired and checked by reading, not compiled, not run on hardware
//
#include <hip/hip_runtime.h>

#define D 200
#define NENT 14541
#define NE 544230
#define HALF (NE/2)
#define NREL2 474
#define BB 2048
#define NF 96
#define FLAT 38400
#define IMGPIX 400

typedef __attribute__((ext_vector_type(16))) _Float16 v16h;
typedef __attribute__((ext_vector_type(8)))  _Float16 v8h;
typedef __attribute__((ext_vector_type(4)))  _Float16 v4h;
typedef __attribute__((ext_vector_type(8)))  float    v8f;

// ---- CDNA5 async Memory->LDS copy (ASYNCcnt), guarded so either toolchain compiles ----
#if defined(__gfx1250__) && __has_builtin(__builtin_amdgcn_global_load_async_to_lds_b128)
#define HAVE_ASYNC_LDS 1
typedef int async_v4i __attribute__((vector_size(4 * sizeof(int))));
#define GPTR(p) ((__attribute__((address_space(1))) async_v4i*)(p))
#define LPTR(p) ((__attribute__((address_space(3))) async_v4i*)(p))
__device__ __forceinline__ void async_wait0() {
#if __has_builtin(__builtin_amdgcn_s_wait_asynccnt)
  __builtin_amdgcn_s_wait_asynccnt(0);
#else
  asm volatile("s_wait_asynccnt 0x0" ::: "memory");
#endif
}
#else
#define HAVE_ASYNC_LDS 0
#endif

// ---- order-preserving float <-> uint encode for atomicMax-based segment max ----
__device__ __forceinline__ unsigned fenc(float f) {
  unsigned b = __float_as_uint(f);
  return b ^ (unsigned)(((int)b >> 31) | 0x80000000);
}
__device__ __forceinline__ float fdec(unsigned u) {
  unsigned b = (u & 0x80000000u) ? (u ^ 0x80000000u) : ~u;
  return __uint_as_float(b);
}

__global__ void k_fill_u32(unsigned* __restrict__ p, long long n, unsigned v) {
  long long i = (long long)blockIdx.x * blockDim.x + threadIdx.x;
  if (i < n) p[i] = v;
}

// =====================================================================
// Generic WMMA GEMM: C[MxN] = op(A)[MxK] @ B[KxN]  (f16 compute, f32 acc)
//   gIdx/gType/gRel : A row = A[gIdx[m]] * gRel[gType[m]]  (edge messages)
//   aScale          : A row scaled elementwise by aScale[k] (loop message)
//   epi: 0 = store, 1 = +bias[col], 2 = +bias[col] then sigmoid
// Block = 256 threads = 8 waves; block tile 128(M) x 32(N); K staged 32/iter.
// LDS tiles are stored k-contiguous (lane half h takes k = 16h..16h+15) so
// each lane reads its 16 halves as one 32B access (2x ds_load_b128).
// =====================================================================
__global__ __launch_bounds__(256) void k_gemm(
    const float* __restrict__ A, const float* __restrict__ Bm, float* __restrict__ C,
    int M, int N, int K, int lda, int ldb, int ldc,
    const int* __restrict__ gIdx, const int* __restrict__ gType,
    const float* __restrict__ gRel, const float* __restrict__ aScale,
    const float* __restrict__ bias, int epi)
{
  // Af[r][*]: row r's 32 k-halves in natural k order; consumer lane reads 16 contiguously.
  // Bf[n][k]: B staged transposed; lane's k-range (16) is contiguous.
  __shared__ __attribute__((aligned(64))) _Float16 Af[128][32];
  __shared__ __attribute__((aligned(64))) _Float16 Bf[32][32];
  const int tid  = threadIdx.x;
  const int wave = tid >> 5, lane = tid & 31;
  const int m0 = blockIdx.x * 128, n0 = blockIdx.y * 32;
  const bool hasG = (gIdx   != nullptr);
  const bool hasR = (gType  != nullptr);
  const bool hasS = (aScale != nullptr);
  v8f acc0 = {}, acc1 = {};

  for (int k0 = 0; k0 < K; k0 += 32) {
    if (k0 + 32 < K)
      __builtin_prefetch((const void*)&Bm[(size_t)(k0 + 32) * (size_t)ldb + n0], 0, 1);

    // ---- stage A: 128x32 halves; each thread does two 8-wide chunks ----
#pragma unroll
    for (int it = 0; it < 2; ++it) {
      int q = tid + (it << 8);            // 0..511
      int r = q >> 2, cc = q & 3;         // row in tile, k-chunk (8 wide)
      int m  = m0 + r;
      int kb = k0 + (cc << 3);
      float vals[8];
#pragma unroll
      for (int j = 0; j < 8; ++j) vals[j] = 0.f;
      if (m < M && kb < K) {
        size_t row = hasG ? (size_t)gIdx[m] : (size_t)m;
        const float* ap = A + row * (size_t)lda + kb;
        if (kb + 8 <= K) {
          float4 u0 = *(const float4*)ap;
          float4 u1 = *(const float4*)(ap + 4);
          vals[0]=u0.x; vals[1]=u0.y; vals[2]=u0.z; vals[3]=u0.w;
          vals[4]=u1.x; vals[5]=u1.y; vals[6]=u1.z; vals[7]=u1.w;
        } else {
#pragma unroll
          for (int j = 0; j < 8; ++j) if (kb + j < K) vals[j] = ap[j];
        }
        if (hasR) {
          const float* rp = gRel + (size_t)gType[m] * (size_t)D + kb;
          if (kb + 8 <= K) {
            float4 r0 = *(const float4*)rp;
            float4 r1 = *(const float4*)(rp + 4);
            vals[0]*=r0.x; vals[1]*=r0.y; vals[2]*=r0.z; vals[3]*=r0.w;
            vals[4]*=r1.x; vals[5]*=r1.y; vals[6]*=r1.z; vals[7]*=r1.w;
          } else {
#pragma unroll
            for (int j = 0; j < 8; ++j) if (kb + j < K) vals[j] *= rp[j];
          }
        }
        if (hasS) {
          const float* sp = aScale + kb;
          if (kb + 8 <= K) {
            float4 s0 = *(const float4*)sp;
            float4 s1 = *(const float4*)(sp + 4);
            vals[0]*=s0.x; vals[1]*=s0.y; vals[2]*=s0.z; vals[3]*=s0.w;
            vals[4]*=s1.x; vals[5]*=s1.y; vals[6]*=s1.z; vals[7]*=s1.w;
          } else {
#pragma unroll
            for (int j = 0; j < 8; ++j) if (kb + j < K) vals[j] *= sp[j];
          }
        }
      }
      v8h h;
#pragma unroll
      for (int j = 0; j < 8; ++j) h[j] = (_Float16)vals[j];
      // chunk c -> half offset: c0->0, c1->8, c2->16, c3->24
      int pos = cc << 3;
      *reinterpret_cast<v8h*>(&Af[r][pos]) = h;        // one ds_store_b128
    }

    // ---- stage B transposed: Bf[n][k], 4 k per thread, one ds_store_b64 ----
    {
      int n  = tid & 31, kq = tid >> 5;
      int gn = n0 + n;
      v4h hb;
#pragma unroll
      for (int j = 0; j < 4; ++j) {
        int gk = k0 + (kq << 2) + j;
        float v = (gk < K && gn < N) ? Bm[(size_t)gk * (size_t)ldb + gn] : 0.f;
        hb[j] = (_Float16)v;
      }
      *reinterpret_cast<v4h*>(&Bf[n][kq << 2]) = hb;
    }
    __syncthreads();

    // ---- fragments: one 32B LDS read each (2x ds_load_b128) ----
    const int hs = (lane >> 4) << 4;   // 0 or 16 halves
    v16h a  = *reinterpret_cast<const v16h*>(&Af[(wave << 4) + (lane & 15)][hs]);
    v16h b0 = *reinterpret_cast<const v16h*>(&Bf[(lane & 15)][hs]);
    v16h b1 = *reinterpret_cast<const v16h*>(&Bf[16 + (lane & 15)][hs]);
    acc0 = __builtin_amdgcn_wmma_f32_16x16x32_f16(false, a, false, b0, (short)0, acc0, false, false);
    acc1 = __builtin_amdgcn_wmma_f32_16x16x32_f16(false, a, false, b1, (short)0, acc1, false, false);
    __syncthreads();
  }

  const int col0 = n0 + (lane & 15);
  const int col1 = col0 + 16;
#pragma unroll
  for (int r = 0; r < 8; ++r) {
    int row = m0 + (wave << 4) + ((lane >> 4) << 3) + r;
    if (row < M) {
      if (col0 < N) {
        float v = acc0[r];
        if (epi >= 1) v += bias[col0];
        if (epi == 2) v = 1.f / (1.f + __expf(-v));
        C[(size_t)row * (size_t)ldc + col0] = v;
      }
      if (col1 < N) {
        float v = acc1[r];
        if (epi >= 1) v += bias[col1];
        if (epi == 2) v = 1.f / (1.f + __expf(-v));
        C[(size_t)row * (size_t)ldc + col1] = v;
      }
    }
  }
}

// ---- attention score: s = leaky_relu(dot(msg_row, att)); seg max via atomicMax ----
__global__ __launch_bounds__(256) void k_att_score(
    const float* __restrict__ msg, const float* __restrict__ att,
    const int* __restrict__ seg, float* __restrict__ sval,
    unsigned* __restrict__ smax, int nE)
{
  int wave = threadIdx.x >> 5, lane = threadIdx.x & 31;
  int e = blockIdx.x * 8 + wave;
  if (e >= nE) return;
  const float* row = msg + (size_t)e * D;
  float s = 0.f;
  for (int k = lane; k < D; k += 32) s += row[k] * att[k];
#pragma unroll
  for (int m = 16; m; m >>= 1) s += __shfl_xor(s, m, 32);
  if (lane == 0) {
    float sc = (s < 0.f) ? 0.2f * s : s;
    sval[e] = sc;
    atomicMax(&smax[seg[e]], fenc(sc));
  }
}

__global__ void k_softmax_exp(const float* __restrict__ sval, const int* __restrict__ seg,
                              const unsigned* __restrict__ smax, float* __restrict__ den,
                              float* __restrict__ exv, int nE)
{
  int e = blockIdx.x * blockDim.x + threadIdx.x;
  if (e >= nE) return;
  int s = seg[e];
  float ex = __expf(sval[e] - fdec(smax[s]));
  exv[e] = ex;
  atomicAdd(&den[s], ex);
}

__global__ void k_agg(const float* __restrict__ msg, const float* __restrict__ exv,
                      const float* __restrict__ den, const int* __restrict__ seg,
                      float* __restrict__ agg, int nE)
{
  long long i = (long long)blockIdx.x * blockDim.x + threadIdx.x;
  if (i >= (long long)nE * D) return;
  int e = (int)(i / D), k = (int)(i % D);
  int s = seg[e];
  float alpha = exv[e] / (den[s] + 1e-16f);
  atomicAdd(&agg[(size_t)s * D + k], msg[i] * alpha);
}

__global__ void k_ent(const float* __restrict__ a, const float* __restrict__ b,
                      const float* __restrict__ c, float* __restrict__ e, long long n)
{
  long long i = (long long)blockIdx.x * blockDim.x + threadIdx.x;
  if (i < n) e[i] = (a[i] + b[i] + c[i]) * (1.f / 3.f);
}

// per-column mean / rstd over M rows (one block per column)
__global__ __launch_bounds__(256) void k_colstats(const float* __restrict__ X, int M, int N,
                                                  float* __restrict__ mean, float* __restrict__ rstd)
{
  __shared__ float ss[256], sq[256];
  int col = blockIdx.x, tid = threadIdx.x;
  float s = 0.f, q = 0.f;
  for (int r = tid; r < M; r += 256) { float v = X[(size_t)r * N + col]; s += v; q += v * v; }
  ss[tid] = s; sq[tid] = q; __syncthreads();
  for (int o = 128; o; o >>= 1) {
    if (tid < o) { ss[tid] += ss[tid + o]; sq[tid] += sq[tid + o]; }
    __syncthreads();
  }
  if (tid == 0) {
    float m = ss[0] / M, var = sq[0] / M - m * m;
    mean[col] = m; rstd[col] = rsqrtf(var + 1e-5f);
  }
}

// mode 0: tanh(bn(x)); mode 1: relu(bn(x))   (in place)
__global__ void k_bn_apply(float* __restrict__ X, const float* __restrict__ mean,
                           const float* __restrict__ rstd, const float* __restrict__ g,
                           const float* __restrict__ b, long long n, int N, int mode)
{
  long long i = (long long)blockIdx.x * blockDim.x + threadIdx.x;
  if (i >= n) return;
  int col = (int)(i % N);
  float v = (X[i] - mean[col]) * rstd[col] * g[col] + b[col];
  X[i] = mode ? fmaxf(v, 0.f) : tanhf(v);
}

__global__ void k_transpose(const float* __restrict__ X, float* __restrict__ XT)
{
  long long i = (long long)blockIdx.x * blockDim.x + threadIdx.x;
  if (i >= (long long)NENT * D) return;
  int r = (int)(i / D), c = (int)(i % D);
  XT[(size_t)c * NENT + r] = X[i];
}

// gather + chequer permutation -> img[b][400]
__global__ void k_comb(const float* __restrict__ ent, const float* __restrict__ relo,
                       const int* __restrict__ sub, const int* __restrict__ rel,
                       const int* __restrict__ perm, float* __restrict__ img)
{
  long long i = (long long)blockIdx.x * blockDim.x + threadIdx.x;
  if (i >= (long long)BB * IMGPIX) return;
  int b = (int)(i / IMGPIX), t = (int)(i % IMGPIX);
  int p = perm[t];
  img[i] = (p < D) ? ent[(size_t)sub[b] * D + p]
                   : relo[(size_t)rel[b] * D + (p - D)];
}

__global__ __launch_bounds__(256) void k_scalar_stats(const float* __restrict__ x, long long n,
                                                      float* __restrict__ stat)
{
  __shared__ float ss[256], sq[256];
  int tid = threadIdx.x;
  float s = 0.f, q = 0.f;
  for (long long i = (long long)blockIdx.x * blockDim.x + tid; i < n;
       i += (long long)gridDim.x * blockDim.x) {
    float v = x[i]; s += v; q += v * v;
  }
  ss[tid] = s; sq[tid] = q; __syncthreads();
  for (int o = 128; o; o >>= 1) {
    if (tid < o) { ss[tid] += ss[tid + o]; sq[tid] += sq[tid + o]; }
    __syncthreads();
  }
  if (tid == 0) { atomicAdd(&stat[0], ss[0]); atomicAdd(&stat[1], sq[0]); }
}

__global__ void k_stat0_fin(float* __restrict__ st, float n)
{
  float m = st[0] / n, var = st[1] / n - m * m;
  st[2] = m; st[3] = rsqrtf(var + 1e-5f);
}

// circular 9x9 conv, 96 filters, one block per image.
// Image row + full filter bank are staged into LDS with CDNA5 async
// Memory->LDS copies (ASYNCcnt) when the toolchain exposes the builtin.
__global__ __launch_bounds__(256) void k_conv(
    const float* __restrict__ img, const float* __restrict__ filt,
    const float* __restrict__ stat0, const float* __restrict__ g0p,
    const float* __restrict__ b0p, float* __restrict__ xconv,
    float* __restrict__ b1s, float* __restrict__ b1q)
{
  __shared__ __attribute__((aligned(16))) float xn[IMGPIX];
  __shared__ __attribute__((aligned(16))) float fw[NF * 81];
  __shared__ float fsum[NF], fsq[NF];
  int tid = threadIdx.x, b = blockIdx.x;
  float m0 = stat0[2], r0 = stat0[3], g0 = g0p[0], bb0 = b0p[0];

#if HAVE_ASYNC_LDS
  // filters: 96*81*4B = 31104B = 1944 x b128 chunks; image row: 100 x b128
  for (int c = tid; c < (NF * 81) / 4; c += 256)
    __builtin_amdgcn_global_load_async_to_lds_b128(GPTR(filt + (size_t)c * 4),
                                                   LPTR(&fw[c * 4]), 0, 0);
  for (int c = tid; c < IMGPIX / 4; c += 256)
    __builtin_amdgcn_global_load_async_to_lds_b128(GPTR(img + (size_t)b * IMGPIX + (size_t)c * 4),
                                                   LPTR(&xn[c * 4]), 0, 0);
  async_wait0();
  __syncthreads();
  for (int i = tid; i < IMGPIX; i += 256)
    xn[i] = (xn[i] - m0) * r0 * g0 + bb0;     // bn0 applied in LDS
#else
  for (int i = tid; i < IMGPIX; i += 256)
    xn[i] = (img[(size_t)b * IMGPIX + i] - m0) * r0 * g0 + bb0;
  for (int i = tid; i < NF * 81; i += 256) fw[i] = filt[i];
#endif
  if (tid < NF) { fsum[tid] = 0.f; fsq[tid] = 0.f; }
  __syncthreads();

  for (int o = tid; o < FLAT; o += 256) {
    int f = o / IMGPIX, pix = o % IMGPIX, i = pix / 20, j = pix % 20;
    const float* fr = &fw[f * 81];
    float acc = 0.f;
#pragma unroll
    for (int di = 0; di < 9; ++di) {
      int i2 = i + di - 4; i2 += (i2 < 0) ? 20 : 0; i2 -= (i2 >= 20) ? 20 : 0;
#pragma unroll
      for (int dj = 0; dj < 9; ++dj) {
        int j2 = j + dj - 4; j2 += (j2 < 0) ? 20 : 0; j2 -= (j2 >= 20) ? 20 : 0;
        acc += xn[i2 * 20 + j2] * fr[di * 9 + dj];
      }
    }
    xconv[(size_t)b * FLAT + o] = acc;
    atomicAdd(&fsum[f], acc);
    atomicAdd(&fsq[f], acc * acc);
  }
  __syncthreads();
  if (tid < NF) { atomicAdd(&b1s[tid], fsum[tid]); atomicAdd(&b1q[tid], fsq[tid]); }
}

__global__ void k_bn1_fin(const float* __restrict__ s, const float* __restrict__ q,
                          float* __restrict__ m1, float* __restrict__ r1)
{
  int f = threadIdx.x;
  if (f < NF) {
    const float n = (float)BB * (float)IMGPIX;
    float m = s[f] / n, var = q[f] / n - m * m;
    m1[f] = m; r1[f] = rsqrtf(var + 1e-5f);
  }
}

__global__ void k_bn1_apply(float* __restrict__ x, const float* __restrict__ m1,
                            const float* __restrict__ r1, const float* __restrict__ g,
                            const float* __restrict__ b, long long n)
{
  long long i = (long long)blockIdx.x * blockDim.x + threadIdx.x;
  if (i >= n) return;
  int f = (int)((i % FLAT) / IMGPIX);
  float v = (x[i] - m1[f]) * r1[f] * g[f] + b[f];
  x[i] = fmaxf(v, 0.f);
}

extern "C" void kernel_launch(void* const* d_in, const int* in_sizes, int n_in,
                              void* d_out, int out_size, void* d_ws, size_t ws_size,
                              hipStream_t stream)
{
  (void)in_sizes; (void)n_in; (void)out_size; (void)ws_size;
  const float* init_embed = (const float*)d_in[0];
  const float* init_rel   = (const float*)d_in[1];
  const float* loop_rel   = (const float*)d_in[2];
  const float* w_in   = (const float*)d_in[3];
  const float* w_out  = (const float*)d_in[4];
  const float* w_loop = (const float*)d_in[5];
  const float* w_rel  = (const float*)d_in[6];
  const float* att_in  = (const float*)d_in[7];
  const float* att_out = (const float*)d_in[8];
  const float* bn_conv_g = (const float*)d_in[9];
  const float* bn_conv_b = (const float*)d_in[10];
  const float* conv_filt = (const float*)d_in[11];
  const float* bn0_g = (const float*)d_in[12];
  const float* bn0_b = (const float*)d_in[13];
  const float* bn1_g = (const float*)d_in[14];
  const float* bn1_b = (const float*)d_in[15];
  const float* fc_w  = (const float*)d_in[16];
  const float* fc_b  = (const float*)d_in[17];
  const float* bn2_g = (const float*)d_in[18];
  const float* bn2_b = (const float*)d_in[19];
  const float* biasv = (const float*)d_in[20];
  const int* eidx  = (const int*)d_in[21];
  const int* etype = (const int*)d_in[22];
  const int* subv  = (const int*)d_in[23];
  const int* relv  = (const int*)d_in[24];
  const int* perm  = (const int*)d_in[25];
  const int* src = eidx;
  const int* dst = eidx + NE;

  char* w = (char*)d_ws;
  auto alloc = [&](size_t bytes) -> void* {
    void* p = (void*)w; w += (bytes + 255) & ~(size_t)255; return p;
  };
  float*    msg   = (float*)alloc((size_t)NE * D * 4);
  float*    sval  = (float*)alloc((size_t)NE * 4);
  float*    exv   = (float*)alloc((size_t)NE * 4);
  unsigned* smax  = (unsigned*)alloc((size_t)2 * NENT * 4);
  float*    den   = (float*)alloc((size_t)2 * NENT * 4);
  float*    agg   = (float*)alloc((size_t)2 * NENT * D * 4);
  float*    loopm = (float*)alloc((size_t)NENT * D * 4);
  float*    ent   = (float*)alloc((size_t)NENT * D * 4);
  float*    entT  = (float*)alloc((size_t)NENT * D * 4);
  float*    relo  = (float*)alloc((size_t)NREL2 * D * 4);
  float*    img   = (float*)alloc((size_t)BB * IMGPIX * 4);
  float*    stat0 = (float*)alloc(4 * 4);
  float*    b1s   = (float*)alloc(NF * 4);
  float*    b1q   = (float*)alloc(NF * 4);
  float*    b1m   = (float*)alloc(NF * 4);
  float*    b1r   = (float*)alloc(NF * 4);
  float*    cmean = (float*)alloc(D * 4);
  float*    crstd = (float*)alloc(D * 4);
  float*    xconv = (float*)alloc((size_t)BB * FLAT * 4);
  float*    xfc   = (float*)alloc((size_t)BB * D * 4);
  float*    out   = (float*)d_out;

  auto cdiv = [](long long a, long long b) { return (unsigned)((a + b - 1) / b); };
  dim3 blk(256);

  // re-init accumulators every call (deterministic across graph replays)
  k_fill_u32<<<cdiv(2 * NENT, 256), blk, 0, stream>>>(smax, 2 * NENT, 0x007FFFFFu); // enc(-inf)
  k_fill_u32<<<cdiv(2 * NENT, 256), blk, 0, stream>>>((unsigned*)den, 2 * NENT, 0u);
  k_fill_u32<<<cdiv((long long)2 * NENT * D, 256), blk, 0, stream>>>((unsigned*)agg, (long long)2 * NENT * D, 0u);
  k_fill_u32<<<1, blk, 0, stream>>>((unsigned*)stat0, 4, 0u);
  k_fill_u32<<<1, blk, 0, stream>>>((unsigned*)b1s, NF, 0u);
  k_fill_u32<<<1, blk, 0, stream>>>((unsigned*)b1q, NF, 0u);

  // rel_out = init_rel @ w_rel
  k_gemm<<<dim3(cdiv(NREL2, 128), cdiv(D, 32)), blk, 0, stream>>>(
      init_rel, w_rel, relo, NREL2, D, D, D, D, D,
      nullptr, nullptr, nullptr, nullptr, nullptr, 0);

  // edge messages: (embed[dst] * rel[etype]) @ w_in / w_out
  k_gemm<<<dim3(cdiv(HALF, 128), cdiv(D, 32)), blk, 0, stream>>>(
      init_embed, w_in, msg, HALF, D, D, D, D, D,
      dst, etype, init_rel, nullptr, nullptr, 0);
  k_gemm<<<dim3(cdiv(HALF, 128), cdiv(D, 32)), blk, 0, stream>>>(
      init_embed, w_out, msg + (size_t)HALF * D, HALF, D, D, D, D, D,
      dst + HALF, etype + HALF, init_rel, nullptr, nullptr, 0);

  // segment softmax aggregation
  k_att_score<<<cdiv(HALF, 8), blk, 0, stream>>>(msg, att_in, src, sval, smax, HALF);
  k_att_score<<<cdiv(HALF, 8), blk, 0, stream>>>(msg + (size_t)HALF * D, att_out,
      src + HALF, sval + HALF, smax + NENT, HALF);
  k_softmax_exp<<<cdiv(HALF, 256), blk, 0, stream>>>(sval, src, smax, den, exv, HALF);
  k_softmax_exp<<<cdiv(HALF, 256), blk, 0, stream>>>(sval + HALF, src + HALF,
      smax + NENT, den + NENT, exv + HALF, HALF);
  k_agg<<<cdiv((long long)HALF * D, 256), blk, 0, stream>>>(msg, exv, den, src, agg, HALF);
  k_agg<<<cdiv((long long)HALF * D, 256), blk, 0, stream>>>(msg + (size_t)HALF * D,
      exv + HALF, den + NENT, src + HALF, agg + (size_t)NENT * D, HALF);

  // loop message: (embed * loop_rel) @ w_loop
  k_gemm<<<dim3(cdiv(NENT, 128), cdiv(D, 32)), blk, 0, stream>>>(
      init_embed, w_loop, loopm, NENT, D, D, D, D, D,
      nullptr, nullptr, nullptr, loop_rel, nullptr, 0);

  // ent = tanh(bn((agg_in + agg_out + loop)/3))
  k_ent<<<cdiv((long long)NENT * D, 256), blk, 0, stream>>>(
      agg, agg + (size_t)NENT * D, loopm, ent, (long long)NENT * D);
  k_colstats<<<D, blk, 0, stream>>>(ent, NENT, D, cmean, crstd);
  k_bn_apply<<<cdiv((long long)NENT * D, 256), blk, 0, stream>>>(
      ent, cmean, crstd, bn_conv_g, bn_conv_b, (long long)NENT * D, D, 0);

  // InteractE decoder
  k_comb<<<cdiv((long long)BB * IMGPIX, 256), blk, 0, stream>>>(ent, relo, subv, relv, perm, img);
  k_scalar_stats<<<512, blk, 0, stream>>>(img, (long long)BB * IMGPIX, stat0);
  k_stat0_fin<<<1, 1, 0, stream>>>(stat0, (float)((long long)BB * IMGPIX));
  k_conv<<<BB, blk, 0, stream>>>(img, conv_filt, stat0, bn0_g, bn0_b, xconv, b1s, b1q);
  k_bn1_fin<<<1, 128, 0, stream>>>(b1s, b1q, b1m, b1r);
  k_bn1_apply<<<cdiv((long long)BB * FLAT, 256), blk, 0, stream>>>(
      xconv, b1m, b1r, bn1_g, bn1_b, (long long)BB * FLAT);

  // fc: x @ fc_w + fc_b ; then bn2 + relu
  k_gemm<<<dim3(cdiv(BB, 128), cdiv(D, 32)), blk, 0, stream>>>(
      xconv, fc_w, xfc, BB, D, FLAT, FLAT, D, D,
      nullptr, nullptr, nullptr, nullptr, fc_b, 1);
  k_colstats<<<D, blk, 0, stream>>>(xfc, BB, D, cmean, crstd);
  k_bn_apply<<<cdiv((long long)BB * D, 256), blk, 0, stream>>>(
      xfc, cmean, crstd, bn2_g, bn2_b, (long long)BB * D, D, 1);

  // scores = sigmoid(x @ ent.T + bias)
  k_transpose<<<cdiv((long long)NENT * D, 256), blk, 0, stream>>>(ent, entT);
  k_gemm<<<dim3(cdiv(BB, 128), cdiv(NENT, 32)), blk, 0, stream>>>(
      xfc, entT, out, BB, NENT, D, D, NENT, NENT,
      nullptr, nullptr, nullptr, nullptr, biasv, 2);
}
